// ImageDiagnostic_20469814132746
// MI455X (gfx1250) — matrix operand order repaired and checked by reading, not compiled, not run on hardware
//
#include <hip/hip_runtime.h>

typedef _Float16 half8 __attribute__((ext_vector_type(8)));
typedef _Float16 v16h  __attribute__((ext_vector_type(16)));
typedef float    v8f   __attribute__((ext_vector_type(8)));

#define BATCH    4
#define NPART    50000
#define MBINS    256
#define KCHUNK   1792                        // multiple of 32
#define NCHUNKS  ((NPART + KCHUNK - 1) / KCHUNK)   // 28
#define IROWS    128                         // i-extent per workgroup (half of output rows)
#define JROWS    256                         // j-extent per workgroup (full)
#define KSTEP    32
#define LSTRIDE  40                          // halfs per LDS row (80 B: 16B-aligned, conflict-free)

__global__ __launch_bounds__(256) void kde_zero(float* __restrict__ out, int n) {
    int i = blockIdx.x * blockDim.x + threadIdx.x;
    if (i < n) out[i] = 0.0f;
}

__global__ __launch_bounds__(256) void kde_partial(const float* __restrict__ x,
                                                   const float* __restrict__ y,
                                                   const float* __restrict__ bins,
                                                   float* __restrict__ out) {
    __shared__ _Float16 sA[IROWS * LSTRIDE];     // kx tiles, [i_row][k] f16
    __shared__ _Float16 sB[JROWS * LSTRIDE];     // ky tiles, [j_row][k] f16
    __shared__ float    sPart[2 * KCHUNK];       // staged particle coords (x then y)

    const int tid  = threadIdx.x;
    const int kc   = blockIdx.x;                 // k-chunk
    const int half = blockIdx.y;                 // output half 0..1
    const int b    = blockIdx.z;                 // batch
    const int qi   = half * IROWS;               // i-bin offset of this half

    const float bw   = bins[1] - bins[0];        // bandwidth = bin resolution
    const float coef = -0.5f / (bw * bw);

    // ---- stage this chunk's particles into LDS (coalesced); OOB -> sentinel giving exp()==0
    const int kbase = kc * KCHUNK;
    for (int i = tid; i < KCHUNK; i += 256) {
        int g = kbase + i;
        float xv = 1e9f, yv = 1e9f;
        if (g < NPART) { xv = x[(size_t)b * NPART + g]; yv = y[(size_t)b * NPART + g]; }
        sPart[i]          = xv;
        sPart[KCHUNK + i] = yv;
    }

    // ---- producer roles: 384 rows per k-step (128 kx + 256 ky); thread t owns rows t and t+256
    const bool  isY0  = (tid >= 128);
    const float binv0 = bins[isY0 ? (tid - 128) : (qi + tid)];
    const float* pSrc0 = isY0 ? (sPart + KCHUNK) : sPart;
    _Float16* rowPtr0 = isY0 ? (sB + (tid - 128) * LSTRIDE) : (sA + tid * LSTRIDE);
    // second row only for tid < 128: ky row tid+128
    const float binv1 = bins[(tid & 127) + 128];
    const float* pSrc1 = sPart + KCHUNK;
    _Float16* rowPtr1 = sB + ((tid & 127) + 128) * LSTRIDE;

    // ---- consumer roles: wave w owns i-tile w (16 rows) x 16 j-tiles
    const int wave = tid >> 5;
    const int lane = tid & 31;
    const int aRow = wave * 16 + (lane & 15);    // row in sA for A fragment
    const int aK0  = (lane >> 4) * 16;           // halfs: lanes0-15 K0..15, lanes16-31 K16..31
    const int bK0  = (lane >> 4) * 16;           // halfs: lanes0-15 K0..15, lanes16-31 K16..31
    const int bRow0 = (lane & 15);               // j-row within tile

    v8f acc[16] = {};                            // 16 j-tiles of 16x16 f32

    __syncthreads();

    const int nsteps = KCHUNK / KSTEP;           // 56
    for (int ks = 0; ks < nsteps; ++ks) {
        // -------- produce: 32 f16 kernel values per row; threads 0-127 produce 2 rows
        {
            const float* p0 = pSrc0 + ks * KSTEP;
            #pragma unroll
            for (int kk = 0; kk < KSTEP; kk += 8) {
                half8 h;
                #pragma unroll
                for (int e = 0; e < 8; ++e) {
                    float d = p0[kk + e] - binv0;
                    h[e] = (_Float16)__expf(coef * d * d);   // v_exp_f32 (TRANS, co-executes w/ XDL)
                }
                *(half8*)(rowPtr0 + kk) = h;
            }
            if (tid < 128) {
                const float* p1 = pSrc1 + ks * KSTEP;
                #pragma unroll
                for (int kk = 0; kk < KSTEP; kk += 8) {
                    half8 h;
                    #pragma unroll
                    for (int e = 0; e < 8; ++e) {
                        float d = p1[kk + e] - binv1;
                        h[e] = (_Float16)__expf(coef * d * d);
                    }
                    *(half8*)(rowPtr1 + kk) = h;
                }
            }
        }
        __syncthreads();

        // -------- consume: A fragment once; 16 WMMAs, B fragments loaded in pairs so each
        //          s_wait_dscnt is amortized over two WMMAs and loads overlap XDL execution
        v16h aF;
        *((half8*)&aF)     = *(const half8*)(sA + aRow * LSTRIDE + aK0);
        *((half8*)&aF + 1) = *(const half8*)(sA + aRow * LSTRIDE + aK0 + 8);

        #pragma unroll
        for (int tp = 0; tp < 8; ++tp) {
            const int r0 = (2 * tp)     * 16 + bRow0;
            const int r1 = (2 * tp + 1) * 16 + bRow0;
            v16h b0, b1;
            *((half8*)&b0)     = *(const half8*)(sB + r0 * LSTRIDE + bK0);
            *((half8*)&b0 + 1) = *(const half8*)(sB + r0 * LSTRIDE + bK0 + 8);
            *((half8*)&b1)     = *(const half8*)(sB + r1 * LSTRIDE + bK0);
            *((half8*)&b1 + 1) = *(const half8*)(sB + r1 * LSTRIDE + bK0 + 8);
            acc[2 * tp]     = __builtin_amdgcn_wmma_f32_16x16x32_f16(
                false, aF, false, b0, (short)0, acc[2 * tp],     false, false);
            acc[2 * tp + 1] = __builtin_amdgcn_wmma_f32_16x16x32_f16(
                false, aF, false, b1, (short)0, acc[2 * tp + 1], false, false);
        }
        __syncthreads();
    }

    // ---- accumulate partial tiles into global output (C layout: M=r+8*(lane>=16), N=lane&15)
    float* outB = out + ((size_t)b << 16);
    const int mBase = qi + wave * 16 + 8 * (lane >> 4);
    const int nBase = (lane & 15);
    #pragma unroll
    for (int tj = 0; tj < 16; ++tj) {
        const int ncol = nBase + tj * 16;
        #pragma unroll
        for (int r = 0; r < 8; ++r) {
            atomicAdd(&outB[(size_t)(mBase + r) * MBINS + ncol], acc[tj][r]);
        }
    }
}

__global__ __launch_bounds__(256) void kde_normalize(float* __restrict__ out) {
    __shared__ float red[256];
    const int b = blockIdx.x;
    float* p = out + ((size_t)b << 16);
    float s = 0.0f;
    for (int i = threadIdx.x; i < MBINS * MBINS; i += 256) s += p[i];
    red[threadIdx.x] = s;
    __syncthreads();
    for (int off = 128; off > 0; off >>= 1) {
        if (threadIdx.x < off) red[threadIdx.x] += red[threadIdx.x + off];
        __syncthreads();
    }
    const float inv = 1.0f / (red[0] + 1e-10f);
    for (int i = threadIdx.x; i < MBINS * MBINS; i += 256) p[i] *= inv;
}

extern "C" void kernel_launch(void* const* d_in, const int* in_sizes, int n_in,
                              void* d_out, int out_size, void* d_ws, size_t ws_size,
                              hipStream_t stream) {
    const float* x    = (const float*)d_in[0];   // (B, N) f32
    const float* y    = (const float*)d_in[1];   // (B, N) f32
    const float* bins = (const float*)d_in[2];   // (M,)   f32
    float* out = (float*)d_out;                  // (B, M, M) f32

    kde_zero<<<(out_size + 255) / 256, 256, 0, stream>>>(out, out_size);

    dim3 grid(NCHUNKS, 2, BATCH);
    kde_partial<<<grid, 256, 0, stream>>>(x, y, bins, out);

    kde_normalize<<<BATCH, 256, 0, stream>>>(out);
}
